// MultiHeadAttention_26268019982844
// MI455X (gfx1250) — compile-verified, matrix operand order checked by reading, not run on hardware
//
#include <hip/hip_runtime.h>

#define B_  4
#define S_  2048
#define D_  1024
#define H_  16
#define DK_ 64
#define M_  (B_ * S_)   // 8192 tokens

typedef __attribute__((ext_vector_type(16))) __bf16 v16bf;
typedef __attribute__((ext_vector_type(8)))  float  v8f;

union FragA { v16bf v; unsigned short us[16]; unsigned int u[8]; };

__device__ __forceinline__ unsigned short f2bf(float f) {
    unsigned int u = __float_as_uint(f);
    unsigned int r = u + 0x7FFFu + ((u >> 16) & 1u);   // round-to-nearest-even
    return (unsigned short)(r >> 16);
}

// group-of-32 xor swizzle (wave32) -- avoids HIP __shfl wavefront-size macro issues
#define SWIZ_IMM(m) ((((m) & 0x1F) << 10) | 0x1F)
#define SWIZF(x, m) __uint_as_float((unsigned)__builtin_amdgcn_ds_swizzle((int)__float_as_uint(x), SWIZ_IMM(m)))

// gfx1250 async global->LDS copy (16B per lane), tracked by ASYNCcnt.
// lds_off = addrspace(3) byte offset (low 32 bits of generic pointer to __shared__).
__device__ __forceinline__ void async_copy_b128(unsigned lds_off, const void* gaddr) {
    asm volatile("global_load_async_to_lds_b128 %0, %1, off"
                 :: "v"(lds_off), "v"(gaddr) : "memory");
}
__device__ __forceinline__ void wait_async0() {
    asm volatile("s_wait_asynccnt 0x0" ::: "memory");
}
__device__ __forceinline__ unsigned lds_off32(const void* p) {
    return (unsigned)(size_t)p;   // generic = {aperture_hi, lds_offset_lo}
}

// Load a 16x32 bf16 fragment (A layout; B layout identical with rows = columns).
// p points at [row][k] row-major halves, `stride` halves/row, k-base pre-applied.
// ISA 7.12.2: lane L holds row row0+(L&15); VGPR i holds K pair at
// 8*(L>>4) + (i<4 ? 2i : 16+2(i-4)).  (contiguous pairs -> compiler merges to b128)
__device__ __forceinline__ v16bf load_frag16(const unsigned short* p, int stride, int row0, int lane) {
    FragA f;
    const unsigned short* rp = p + (size_t)(row0 + (lane & 15)) * stride;
    const int hb = (lane >> 4) * 8;
#pragma unroll
    for (int i = 0; i < 8; ++i) {
        int kk = hb + ((i < 4) ? (2 * i) : (16 + 2 * (i - 4)));
        f.u[i] = *(const unsigned int*)(rp + kk);
    }
    return f.v;
}

__global__ void cvt_f32_bf16(const float* __restrict__ in, unsigned short* __restrict__ out, int n) {
    int i = blockIdx.x * 256 + threadIdx.x;
    if (i < n) out[i] = f2bf(in[i]);
}

// Transpose + convert: out[n*D_ + k] = bf16(in[k*D_ + n]); 32x32 LDS tiles.
__global__ __launch_bounds__(256) void cvt_transpose_bf16(const float* __restrict__ in,
                                                          unsigned short* __restrict__ out) {
    __shared__ float tile[32][33];
    const int k0 = blockIdx.x * 32, n0 = blockIdx.y * 32;
    const int tx = threadIdx.x & 31, ty = threadIdx.x >> 5;   // 32 x 8
#pragma unroll
    for (int r = 0; r < 32; r += 8)
        tile[ty + r][tx] = in[(size_t)(k0 + ty + r) * D_ + n0 + tx];   // tile[k][n]
    __syncthreads();
#pragma unroll
    for (int r = 0; r < 32; r += 8)
        out[(size_t)(n0 + ty + r) * D_ + k0 + tx] = f2bf(tile[tx][ty + r]);
}

// C = A[M_ x D_] * W + bias; W given TRANSPOSED: Wt[n][k]. bf16 WMMA, f32 accum.
// Block tile 128x128, 8 waves of 32x64, K-step 32, double-buffered LDS filled by
// gfx1250 async global->LDS copies (ASYNCcnt), one barrier per K-step.
// mode 0: bf16 out, [B,H,S,DK] layout (Q,K), val=(acc+bias)*scale
// mode 1: f32 out, [M_,D_] (final projection)
// mode 2: bf16 out, [B,H,DK,S] layout (V transposed for fast PV fragments)
__global__ __launch_bounds__(256) void gemm_qkv(
    const unsigned short* __restrict__ A,
    const unsigned short* __restrict__ Wt,
    const float* __restrict__ bias,
    unsigned short* __restrict__ outb,
    float* __restrict__ outf,
    float scale, int mode)
{
    __shared__ unsigned short As[2][128][40];   // [buf][m][k], padded
    __shared__ unsigned short Bs[2][128][40];   // [buf][n][k], padded

    const int t    = threadIdx.x;
    const int lane = t & 31;
    const int wid  = t >> 5;
    const int m0   = blockIdx.y * 128;
    const int n0   = blockIdx.x * 128;
    const int wm   = (wid & 3) * 32;         // 4 wave rows
    const int wn   = (wid >> 2) * 64;        // 2 wave cols

    v8f acc[2][4];
#pragma unroll
    for (int mi = 0; mi < 2; ++mi)
#pragma unroll
        for (int ni = 0; ni < 4; ++ni) acc[mi][ni] = (v8f){};

    const int srow = t >> 1, soff = (t & 1) * 16;   // staging: 32B per thread per tile
    const unsigned short* gA = A  + (size_t)(m0 + srow) * D_ + soff;
    const unsigned short* gB = Wt + (size_t)(n0 + srow) * D_ + soff;

    // prologue: async-stage k=0 into buffer 0
    async_copy_b128(lds_off32(&As[0][srow][soff]),     gA);
    async_copy_b128(lds_off32(&As[0][srow][soff + 8]), gA + 8);
    async_copy_b128(lds_off32(&Bs[0][srow][soff]),     gB);
    async_copy_b128(lds_off32(&Bs[0][srow][soff + 8]), gB + 8);

    for (int k0 = 0; k0 < D_; k0 += 32) {
        const int p = (k0 >> 5) & 1;
        wait_async0();          // own transfers into buf p are done
        __syncthreads();        // ...and everyone's; buf p^1 free for refill

        if (k0 + 32 < D_) {     // kick off next tile while computing this one
            const unsigned short* nA = gA + k0 + 32;
            const unsigned short* nB = gB + k0 + 32;
            async_copy_b128(lds_off32(&As[p ^ 1][srow][soff]),     nA);
            async_copy_b128(lds_off32(&As[p ^ 1][srow][soff + 8]), nA + 8);
            async_copy_b128(lds_off32(&Bs[p ^ 1][srow][soff]),     nB);
            async_copy_b128(lds_off32(&Bs[p ^ 1][srow][soff + 8]), nB + 8);
        }

        v16bf a0 = load_frag16(&As[p][0][0], 40, wm,      lane);
        v16bf a1 = load_frag16(&As[p][0][0], 40, wm + 16, lane);
#pragma unroll
        for (int ni = 0; ni < 4; ++ni) {
            v16bf b = load_frag16(&Bs[p][0][0], 40, wn + ni * 16, lane);
            acc[0][ni] = __builtin_amdgcn_wmma_f32_16x16x32_bf16(false, a0, false, b, (short)0, acc[0][ni], false, false);
            acc[1][ni] = __builtin_amdgcn_wmma_f32_16x16x32_bf16(false, a1, false, b, (short)0, acc[1][ni], false, false);
        }
    }

    const int col = lane & 15, half = lane >> 4;
#pragma unroll
    for (int mi = 0; mi < 2; ++mi) {
#pragma unroll
        for (int ni = 0; ni < 4; ++ni) {
#pragma unroll
            for (int i = 0; i < 8; ++i) {
                int m = m0 + wm + mi * 16 + i + 8 * half;
                int n = n0 + wn + ni * 16 + col;
                float val = (acc[mi][ni][i] + bias[n]) * scale;
                if (mode == 1) {
                    outf[(size_t)m * D_ + n] = val;
                } else {
                    int b = m / S_, s = m % S_;
                    int h = n >> 6, dk = n & 63;
                    if (mode == 0)
                        outb[((size_t)(b * H_ + h) * S_ + s) * DK_ + dk] = f2bf(val);
                    else // mode 2: V transposed [B,H,DK,S]
                        outb[((size_t)(b * H_ + h) * DK_ + dk) * S_ + s] = f2bf(val);
                }
            }
        }
    }
}

// Flash-attention: 1 wave per 16-query tile, 32-key steps, causal.
// Q pre-scaled by 1/sqrt(DK). Q/K: [B*H][S][DK] bf16; V: [B*H][DK][S] bf16 (transposed).
// Output [B*S][D] bf16.
__global__ __launch_bounds__(256) void attn_kernel(
    const unsigned short* __restrict__ Q,
    const unsigned short* __restrict__ Km,
    const unsigned short* __restrict__ Vt,
    unsigned short* __restrict__ Ob)
{
    __shared__ unsigned short Ps[8][16][36];  // per-wave P tile [q][key], padded

    const int t = threadIdx.x, lane = t & 31, w = t >> 5;
    const int gid = blockIdx.x * 8 + w;       // 8192 q-tiles total
    const int bh  = gid >> 7;                 // 0..63
    const int qt  = gid & 127;
    const int q0  = qt * 16;
    const int b   = bh >> 4, h = bh & 15;
    const size_t base = (size_t)bh * S_ * DK_;
    const unsigned short* Qb  = Q  + base;
    const unsigned short* Kb  = Km + base;
    const unsigned short* Vtb = Vt + base;    // [DK][S]

    const int col = lane & 15, half = lane >> 4;

    v16bf qa0 = load_frag16(Qb,      DK_, q0, lane);   // dk 0..31
    v16bf qa1 = load_frag16(Qb + 32, DK_, q0, lane);   // dk 32..63

    v8f acc0 = {}, acc1 = {}, acc2 = {}, acc3 = {};
    float mrow[8], lrow[8];
#pragma unroll
    for (int i = 0; i < 8; ++i) { mrow[i] = -3.0e38f; lrow[i] = 0.0f; }

    const int nsteps = (q0 + 16 + 31) >> 5;

    for (int j = 0; j < nsteps; ++j) {
        const int kb = j * 32;

        // S tile 16x32 = two 16x16 column groups, each K-split over dk
        v8f s0 = {}, s1 = {};
        v16bf k00 = load_frag16(Kb,      DK_, kb,      lane);
        v16bf k01 = load_frag16(Kb + 32, DK_, kb,      lane);
        s0 = __builtin_amdgcn_wmma_f32_16x16x32_bf16(false, qa0, false, k00, (short)0, s0, false, false);
        s0 = __builtin_amdgcn_wmma_f32_16x16x32_bf16(false, qa1, false, k01, (short)0, s0, false, false);
        v16bf k10 = load_frag16(Kb,      DK_, kb + 16, lane);
        v16bf k11 = load_frag16(Kb + 32, DK_, kb + 16, lane);
        s1 = __builtin_amdgcn_wmma_f32_16x16x32_bf16(false, qa0, false, k10, (short)0, s1, false, false);
        s1 = __builtin_amdgcn_wmma_f32_16x16x32_bf16(false, qa1, false, k11, (short)0, s1, false, false);

        // causal mask + online softmax per row (row = i + 8*half; col = key lane)
#pragma unroll
        for (int i = 0; i < 8; ++i) {
            const int qg = q0 + i + 8 * half;
            float v0 = ((kb + col)      <= qg) ? s0[i] : -3.0e38f;
            float v1 = ((kb + 16 + col) <= qg) ? s1[i] : -3.0e38f;
            float mx = fmaxf(v0, v1);
            mx = fmaxf(mx, SWIZF(mx, 1)); mx = fmaxf(mx, SWIZF(mx, 2));
            mx = fmaxf(mx, SWIZF(mx, 4)); mx = fmaxf(mx, SWIZF(mx, 8));
            float mnew = fmaxf(mrow[i], mx);
            float p0 = __expf(v0 - mnew);
            float p1 = __expf(v1 - mnew);
            float rs = p0 + p1;
            rs += SWIZF(rs, 1); rs += SWIZF(rs, 2); rs += SWIZF(rs, 4); rs += SWIZF(rs, 8);
            float fcorr = __expf(mrow[i] - mnew);
            lrow[i] = lrow[i] * fcorr + rs;
            mrow[i] = mnew;
            acc0[i] *= fcorr; acc1[i] *= fcorr; acc2[i] *= fcorr; acc3[i] *= fcorr;
            Ps[w][i + 8 * half][col]      = f2bf(p0);
            Ps[w][i + 8 * half][col + 16] = f2bf(p1);
        }
        asm volatile("s_wait_dscnt 0x0" ::: "memory");   // intra-wave LDS RAW before cross-lane read

        v16bf pa = load_frag16(&Ps[w][0][0], 36, 0, lane);   // P as A fragment 16x32

        // O += P(16x32) @ V(32x64); V transposed -> contiguous b128 fragment loads
#pragma unroll
        for (int tt = 0; tt < 4; ++tt) {
            v16bf vb = load_frag16(Vtb + kb, S_, tt * 16, lane);
            if      (tt == 0) acc0 = __builtin_amdgcn_wmma_f32_16x16x32_bf16(false, pa, false, vb, (short)0, acc0, false, false);
            else if (tt == 1) acc1 = __builtin_amdgcn_wmma_f32_16x16x32_bf16(false, pa, false, vb, (short)0, acc1, false, false);
            else if (tt == 2) acc2 = __builtin_amdgcn_wmma_f32_16x16x32_bf16(false, pa, false, vb, (short)0, acc2, false, false);
            else              acc3 = __builtin_amdgcn_wmma_f32_16x16x32_bf16(false, pa, false, vb, (short)0, acc3, false, false);
        }
    }

    // finalize: O /= l, write bf16 [B,S,D] (heads re-interleaved)
#pragma unroll
    for (int i = 0; i < 8; ++i) {
        float inv = 1.0f / lrow[i];
        int qg = q0 + i + 8 * half;
        size_t rowbase = ((size_t)(b * S_ + qg)) * D_ + h * DK_;
        Ob[rowbase +  0 + col] = f2bf(acc0[i] * inv);
        Ob[rowbase + 16 + col] = f2bf(acc1[i] * inv);
        Ob[rowbase + 32 + col] = f2bf(acc2[i] * inv);
        Ob[rowbase + 48 + col] = f2bf(acc3[i] * inv);
    }
}

extern "C" void kernel_launch(void* const* d_in, const int* in_sizes, int n_in,
                              void* d_out, int out_size, void* d_ws, size_t ws_size,
                              hipStream_t stream) {
    (void)in_sizes; (void)n_in; (void)out_size; (void)ws_size;
    const float* x  = (const float*)d_in[0];
    const float* wq = (const float*)d_in[1];
    const float* bq = (const float*)d_in[2];
    const float* wk = (const float*)d_in[3];
    const float* bk = (const float*)d_in[4];
    const float* wv = (const float*)d_in[5];
    const float* bv = (const float*)d_in[6];
    const float* wo = (const float*)d_in[7];
    const float* bo = (const float*)d_in[8];
    // d_in[9] = causal mask, handled analytically
    float* out = (float*)d_out;

    char* ws = (char*)d_ws;
    const size_t MB = 1024 * 1024;
    unsigned short* xb  = (unsigned short*)(ws);             // 16 MB: x bf16 [8192][1024]
    unsigned short* wqb = (unsigned short*)(ws + 16 * MB);   // 2 MB each, transposed [n][k]
    unsigned short* wkb = (unsigned short*)(ws + 18 * MB);
    unsigned short* wvb = (unsigned short*)(ws + 20 * MB);
    unsigned short* wob = (unsigned short*)(ws + 22 * MB);
    unsigned short* Qw  = (unsigned short*)(ws + 24 * MB);   // 16 MB each
    unsigned short* Kw  = (unsigned short*)(ws + 40 * MB);
    unsigned short* Vw  = (unsigned short*)(ws + 56 * MB);   // [B,H,DK,S]
    unsigned short* Obf = (unsigned short*)(ws + 72 * MB);   // 16 MB, attn out bf16 [8192][1024]

    cvt_f32_bf16<<<(M_ * D_ + 255) / 256, 256, 0, stream>>>(x, xb, M_ * D_);
    dim3 gt(D_ / 32, D_ / 32, 1);
    cvt_transpose_bf16<<<gt, 256, 0, stream>>>(wq, wqb);
    cvt_transpose_bf16<<<gt, 256, 0, stream>>>(wk, wkb);
    cvt_transpose_bf16<<<gt, 256, 0, stream>>>(wv, wvb);
    cvt_transpose_bf16<<<gt, 256, 0, stream>>>(wo, wob);

    dim3 g(D_ / 128, M_ / 128, 1);   // (8, 64)
    gemm_qkv<<<g, 256, 0, stream>>>(xb, wqb, bq, Qw, nullptr, 0.125f, 0); // 1/sqrt(64)
    gemm_qkv<<<g, 256, 0, stream>>>(xb, wkb, bk, Kw, nullptr, 1.0f,   0);
    gemm_qkv<<<g, 256, 0, stream>>>(xb, wvb, bv, Vw, nullptr, 1.0f,   2); // V transposed

    attn_kernel<<<(B_ * H_ * (S_ / 16)) / 8, 256, 0, stream>>>(Qw, Kw, Vw, Obf);

    gemm_qkv<<<g, 256, 0, stream>>>(Obf, wob, bo, nullptr, out, 1.0f, 1);
}
